// NequIPConv_18038862643914
// MI455X (gfx1250) — matrix operand order repaired and checked by reading, not compiled, not run on hardware
//
#include <hip/hip_runtime.h>
#include <hip/hip_bf16.h>

// ---------------------------------------------------------------------------
// NequIP edge-conv for MI455X (gfx1250, wave32, WMMA bf16 16x16x32)
//
//   msg1[e,o] = SiLU( b1[o] + sum_{c,r} x[col[e],c] * rbf[e,r] * M[c*8+r,o] )
//   msg2[e,o] = b2[o] + msg1[e,:] @ W2
//   out       = x + scatter_add(msg2, row)
//
// with M[c*8+r,o] = sum_{o'} W_tp[c,r,o'] * W1[r*32+o',o]  (fused, 256x32)
// ---------------------------------------------------------------------------

#define NN_NODES 50000
#define NN_EDGES 800000
#define CC 32
#define RR 8

typedef __bf16 bf16_t;
typedef bf16_t v16bf __attribute__((ext_vector_type(16)));
typedef float  v8f   __attribute__((ext_vector_type(8)));
typedef unsigned int v4u  __attribute__((ext_vector_type(4)));
typedef unsigned int u32x4 __attribute__((ext_vector_type(4)));
typedef int          i32x8 __attribute__((ext_vector_type(8)));
typedef int          i32x4 __attribute__((ext_vector_type(4)));

// --- packed f32x2 -> bf16x2 (one dword) ------------------------------------
__device__ __forceinline__ unsigned int pack_bf16(float lo, float hi) {
#if __has_builtin(__builtin_amdgcn_cvt_pk_bf16_f32)
    auto pk = __builtin_amdgcn_cvt_pk_bf16_f32(lo, hi);   // v_cvt_pk_bf16_f32
    return __builtin_bit_cast(unsigned int, pk);
#else
    // round-half-up then pack both high halves with one v_perm_b32
    unsigned int ulo = __builtin_bit_cast(unsigned int, lo) + 0x8000u;
    unsigned int uhi = __builtin_bit_cast(unsigned int, hi) + 0x8000u;
    return __builtin_amdgcn_perm(uhi, ulo, 0x07060302);   // {uhi.hi, ulo.hi}
#endif
}

__device__ __forceinline__ unsigned short f2bf(float f) {   // setup kernel only
    unsigned int u = __builtin_bit_cast(unsigned int, f);
    u += 0x7FFFu + ((u >> 16) & 1u);                        // RNE
    return (unsigned short)(u >> 16);
}

// ---------------------------------------------------------------------------
// Setup: fused M = W_tp x W1 (256x32) and W2 (32x32), stored in WMMA B-matrix
// lane layout (bf16) in d_ws.
//   B layout (16x16x32 bf16): lane = half*16 + n ; element i<8 -> k = half*8+i, i>=8 -> k = 16+half*8+(i-8)
//   M  fragments: ushort[(t*2+j)*512 + lane*16 + i],  t = k-chunk, j = n-tile
//   W2 fragments: ushort[8192 + j*512 + lane*16 + i]
// ---------------------------------------------------------------------------
__global__ void nequip_prep_weights(const float* __restrict__ W_tp,
                                    const float* __restrict__ W1,
                                    const float* __restrict__ W2,
                                    unsigned short* __restrict__ wperm) {
    int tid = blockIdx.x * blockDim.x + threadIdx.x;
    if (tid < 256 * 32) {
        int k = tid >> 5;            // global k = c*8 + r
        int o = tid & 31;
        int c = k >> 3, r = k & 7;
        float s = 0.f;
        #pragma unroll 8
        for (int op = 0; op < 32; ++op)
            s += W_tp[(c * 8 + r) * 32 + op] * W1[(r * 32 + op) * 32 + o];
        int t = k >> 5, kk = k & 31;
        int half = (kk >> 3) & 1, i = (kk & 7) | ((kk >> 4) << 3);
        int j = o >> 4, nn = o & 15;
        int lane = half * 16 + nn;
        wperm[(t * 2 + j) * 512 + lane * 16 + i] = f2bf(s);
    } else if (tid < 256 * 32 + 32 * 32) {
        int id = tid - 256 * 32;
        int k = id >> 5, o = id & 31;
        int half = (k >> 3) & 1, i = (k & 7) | ((k >> 4) << 3);
        int j = o >> 4, nn = o & 15;
        int lane = half * 16 + nn;
        wperm[8192 + j * 512 + lane * 16 + i] = f2bf(W2[k * 32 + o]);
    }
}

// out = x   (residual base; messages are atomically accumulated on top)
__global__ void nequip_init_out(const float* __restrict__ x,
                                float* __restrict__ out, int n) {
    int i = blockIdx.x * blockDim.x + threadIdx.x;
    if (i < n) out[i] = x[i];
}

// ---------------------------------------------------------------------------
// Main kernel: one wave32 per 16-edge tile (grid-strided).
// ---------------------------------------------------------------------------
__launch_bounds__(256, 1)
__global__ void nequip_conv(const float* __restrict__ x,
                            const float* __restrict__ pos,
                            const long long* __restrict__ ei,   // (2, E) int64
                            const float* __restrict__ b1,
                            const float* __restrict__ b2,
                            const unsigned short* __restrict__ wperm,
                            float* __restrict__ out) {
    __shared__ unsigned int   s_w[4608];        // 18KB: fused-M + W2 (bf16, permuted)
    __shared__ unsigned short s_msg1[8][512];   // per-wave 16x32 bf16 transpose buf

    const int tid  = threadIdx.x;
    const int lane = tid & 31;
    const int wv   = tid >> 5;       // wave id in block (0..7)

    // ---- stage permuted weights (18KB) into LDS via the Tensor Data Mover ----
#if __has_builtin(__builtin_amdgcn_tensor_load_to_lds)
    if (wv == 0) {
        const unsigned long long ga = (unsigned long long)(uintptr_t)wperm;
        const unsigned int lds_addr = (unsigned int)(uintptr_t)&s_w[0]; // LDS aperture low bits
        // D# group0: count=1, lds_addr, global_addr, type=2
        u32x4 g0;
        g0[0] = 1u;
        g0[1] = lds_addr;
        g0[2] = (unsigned int)(ga & 0xFFFFFFFFull);
        g0[3] = (unsigned int)((ga >> 32) & 0x01FFFFFFull) | 0x80000000u;
        // D# group1: data_size=4B, 1-row tensor: dim0 = tile_dim0 = 4608 dwords
        i32x8 g1;
        g1[0] = (int)(2u << 16);          // data_size = 2 (4 bytes)
        g1[1] = (int)(4608u << 16);       // tensor_dim0[15:0] in [31:16]
        g1[2] = (int)(1u << 16);          // tensor_dim0[31:16]=0 | tensor_dim1=1
        g1[3] = (int)(4608u << 16);       // tile_dim0 = 4608
        g1[4] = 1;                        // tile_dim1 = 1, tile_dim2 = 0
        g1[5] = 4608;                     // tensor_dim0_stride[31:0]
        g1[6] = (int)(4608u << 16);       // tensor_dim1_stride[15:0]
        g1[7] = 0;
        i32x4 gz4 = {0, 0, 0, 0};
        i32x8 gz8 = {0, 0, 0, 0, 0, 0, 0, 0};
        // 6-arg form (clang-23 / therock-10.0 headers)
        __builtin_amdgcn_tensor_load_to_lds(g0, g1, gz4, gz4, gz8, 0);
        __builtin_amdgcn_s_wait_tensorcnt(0);                     // s_wait_tensorcnt 0
    }
#else
    {
        const unsigned int* wsrc = (const unsigned int*)wperm;
        for (int i = tid; i < 4608; i += 256) s_w[i] = wsrc[i];
    }
#endif
    __syncthreads();

    // ---- hoist all B fragments (tile-invariant) into registers ----
    v16bf Bf[16];
    v16bf W2f[2];
    #pragma unroll
    for (int f = 0; f < 16; ++f) {
        union { v4u q[2]; v16bf v; } u;
        u.q[0] = *(const v4u*)&s_w[f * 256 + lane * 8];
        u.q[1] = *(const v4u*)&s_w[f * 256 + lane * 8 + 4];
        Bf[f] = u.v;
    }
    #pragma unroll
    for (int f = 0; f < 2; ++f) {
        union { v4u q[2]; v16bf v; } u;
        u.q[0] = *(const v4u*)&s_w[4096 + f * 256 + lane * 8];
        u.q[1] = *(const v4u*)&s_w[4096 + f * 256 + lane * 8 + 4];
        W2f[f] = u.v;
    }

    const int m    = lane & 15;      // edge-in-tile (A row) / n-in-tile (C col)
    const int half = lane >> 4;

    const int gwave  = blockIdx.x * 8 + wv;
    const int nwaves = gridDim.x * 8;
    const int ntiles = (NN_EDGES + 15) >> 4;

    const float b1n0 = b1[m];
    const float b1n1 = b1[m + 16];
    const float b2n0 = b2[m];
    const float b2n1 = b2[m + 16];

    for (int tile = gwave; tile < ntiles; tile += nwaves) {
        const int ebase = tile * 16;
        int e = ebase + m; if (e >= NN_EDGES) e = NN_EDGES - 1;   // clamp: EXEC stays full
        const long long rown = ei[e];
        const long long coln = ei[NN_EDGES + e];

        // prefetch next tile's edge indices (global_prefetch_b8)
        if (tile + nwaves < ntiles) {
            const int en = (tile + nwaves) * 16 + m;
            __builtin_prefetch(&ei[en], 0, 3);
            __builtin_prefetch(&ei[NN_EDGES + en], 0, 3);
        }

        // ---- edge geometry -> 8 Gaussian RBFs (both half-lanes duplicate) ----
        const float dx = pos[rown * 3 + 0] - pos[coln * 3 + 0];
        const float dy = pos[rown * 3 + 1] - pos[coln * 3 + 1];
        const float dz = pos[rown * 3 + 2] - pos[coln * 3 + 2];
        const float dist = sqrtf(dx * dx + dy * dy + dz * dz + 1e-12f);
        float rbf[8];
        #pragma unroll
        for (int r = 0; r < 8; ++r) {
            const float d = dist - (5.0f / 7.0f) * (float)r;
            // coeff = -0.5/width^2 (width = 5/7); scaling = 1/sqrt(2*pi)
            rbf[r] = 0.3989422804014327f * __expf(-0.98f * d * d);
        }

        // ---- GEMM1: (16 x 256) outer-product features @ M(256x32), 8 k-chunks
        // A layout: element i<8 -> k = half*8+i  (channel c = 4t+half)
        //           element i>=8 -> k = 16+half*8+(i-8) (channel c = 4t+2+half)
        v8f acc0 = {}; v8f acc1 = {};
        const float* xrow = x + (size_t)coln * CC + half;   // fold `half` into addr
        #pragma unroll
        for (int t = 0; t < 8; ++t) {
            const float xa = xrow[t * 4];         // channel 4t + half
            const float xb = xrow[t * 4 + 2];     // channel 4t + 2 + half
            union { unsigned int d[8]; v16bf v; } af;
            #pragma unroll
            for (int p = 0; p < 4; ++p) {
                af.d[p]     = pack_bf16(xa * rbf[2 * p], xa * rbf[2 * p + 1]);
                af.d[p + 4] = pack_bf16(xb * rbf[2 * p], xb * rbf[2 * p + 1]);
            }
            acc0 = __builtin_amdgcn_wmma_f32_16x16x32_bf16(false, af.v, false, Bf[t * 2 + 0],
                                                           (short)0, acc0, false, false);
            acc1 = __builtin_amdgcn_wmma_f32_16x16x32_bf16(false, af.v, false, Bf[t * 2 + 1],
                                                           (short)0, acc1, false, false);
        }

        // ---- bias + SiLU, stage (C-layout -> A-layout) through per-wave LDS ----
        #pragma unroll
        for (int v = 0; v < 8; ++v) {
            const int row = v + 8 * half;          // edge-in-tile for this acc slot
            float h0 = acc0[v] + b1n0;
            float h1 = acc1[v] + b1n1;
            h0 = h0 * __builtin_amdgcn_rcpf(1.0f + __expf(-h0));   // SiLU
            h1 = h1 * __builtin_amdgcn_rcpf(1.0f + __expf(-h1));
            const unsigned int pk = pack_bf16(h0, h1);
            s_msg1[wv][row * 32 + m]      = (unsigned short)pk;        // b16 store (lo)
            s_msg1[wv][row * 32 + m + 16] = (unsigned short)(pk >> 16);
        }
        asm volatile("s_wait_dscnt 0" ::: "memory");   // in-wave LDS RAW

        // ---- GEMM2: msg1(16x32) @ W2(32x32) ----
        union { v4u q[2]; v16bf v; } a2;
        a2.q[0] = *(const v4u*)&s_msg1[wv][m * 32 + half * 8];
        a2.q[1] = *(const v4u*)&s_msg1[wv][m * 32 + 16 + half * 8];
        v8f o0 = {}; v8f o1 = {};
        o0 = __builtin_amdgcn_wmma_f32_16x16x32_bf16(false, a2.v, false, W2f[0],
                                                     (short)0, o0, false, false);
        o1 = __builtin_amdgcn_wmma_f32_16x16x32_bf16(false, a2.v, false, W2f[1],
                                                     (short)0, o1, false, false);

        // ---- scatter-add messages to target nodes (L2-resident atomics) ----
        #pragma unroll
        for (int v = 0; v < 8; ++v) {
            const int row = v + 8 * half;
            const int node = __shfl((int)rown, row, 32);  // lane `row` holds that edge
            if (ebase + row < NN_EDGES) {
                unsafeAtomicAdd(&out[node * CC + m],      o0[v] + b2n0);
                unsafeAtomicAdd(&out[node * CC + m + 16], o1[v] + b2n1);
            }
        }
    }
}

extern "C" void kernel_launch(void* const* d_in, const int* in_sizes, int n_in,
                              void* d_out, int out_size, void* d_ws, size_t ws_size,
                              hipStream_t stream) {
    const float*     x    = (const float*)d_in[0];
    const float*     pos  = (const float*)d_in[1];
    const long long* ei   = (const long long*)d_in[2];
    const float*     W_tp = (const float*)d_in[3];
    const float*     W1   = (const float*)d_in[4];
    const float*     b1   = (const float*)d_in[5];
    const float*     W2   = (const float*)d_in[6];
    const float*     b2   = (const float*)d_in[7];
    float*           out  = (float*)d_out;
    unsigned short*  wperm = (unsigned short*)d_ws;   // 18 KB used

    // 1) fuse + permute weights into WMMA B layout
    nequip_prep_weights<<<36, 256, 0, stream>>>(W_tp, W1, W2, wperm);

    // 2) out = x (residual base)
    const int n_out = NN_NODES * CC;
    nequip_init_out<<<(n_out + 255) / 256, 256, 0, stream>>>(x, out, n_out);

    // 3) main edge kernel: 625 blocks x 8 waves -> 5000 waves, 10 tiles each
    nequip_conv<<<625, 256, 0, stream>>>(x, pos, ei, b1, b2, wperm, out);
}
